// ErnieLayoutSelfAttention_9251359556005
// MI455X (gfx1250) — matrix operand order repaired and checked by reading, not compiled, not run on hardware
//
#include <hip/hip_runtime.h>
#include <hip/hip_bf16.h>

// ---------------------------------------------------------------------------
// ErnieLayout self-attention for MI455X (gfx1250, wave32, WMMA).
// Kernel 1: fused QKV projection (bf16 WMMA, f32 accum) -> q,k,[v^T] in d_ws.
// Kernel 2: flash-attention with additive rel biases + mask (bf16 WMMA).
// Bandwidth floor: rel_pos + rel_2d_pos + mask streams (~840 MB) dominate.
// ---------------------------------------------------------------------------

#define BB   2
#define SS   2048
#define HIDD 768
#define NHH  12
#define HDD  64
#define FMIN (-3.4028234663852886e38f)

typedef __attribute__((ext_vector_type(16))) __bf16 v16bf;
typedef __attribute__((ext_vector_type(8)))  __bf16 v8bf;
typedef __attribute__((ext_vector_type(8)))  float  v8f;
typedef __attribute__((ext_vector_type(4)))  float  v4f;

__device__ __forceinline__ v16bf cat8(v8bf lo, v8bf hi) {
  return __builtin_shufflevector(lo, hi, 0,1,2,3,4,5,6,7,8,9,10,11,12,13,14,15);
}

// ---------------------------------------------------------------------------
// Kernel 1: q = (hs @ Wq^T + bq) / 8 ; k = hs @ Wk^T + bk ; vT = (hs @ Wv^T + bv)^T
// One wave computes one 16x16 tile of the [4096 x 768] output.
// A (hidden) 16x32 bf16 layout: lane half0 -> K 0..15, half1 -> K 16..31.
// B (W row-major, out[m,n]=sum_k hs[m,k]*W[n,k]) 32x16: lane col=n, K = half*16..+15.
// ---------------------------------------------------------------------------
__global__ __launch_bounds__(256) void qkv_proj_kernel(
    const float* __restrict__ hs,
    const float* __restrict__ Wq, const float* __restrict__ bq,
    const float* __restrict__ Wk, const float* __restrict__ bk,
    const float* __restrict__ Wv, const float* __restrict__ bv,
    __bf16* __restrict__ qb, __bf16* __restrict__ kbuf, __bf16* __restrict__ vT)
{
  const int lane = threadIdx.x & 31;
  const int wave = threadIdx.x >> 5;
  const int half = lane >> 4;
  const int l16  = lane & 15;

  const int m0    = blockIdx.x * 16;
  const int n0    = (blockIdx.y * 8 + wave) * 16;
  const int which = blockIdx.z;                 // 0=q 1=k 2=v

  const float* W    = (which == 0) ? Wq : (which == 1) ? Wk : Wv;
  const float* bias = (which == 0) ? bq : (which == 1) ? bk : bv;

  const int mA = m0 + l16;
  const int nB = n0 + l16;

  v8f acc = {};
  for (int kk = 0; kk < HIDD; kk += 32) {
    // --- A tile (f32 -> bf16) ---
    const float* ap = hs + (size_t)mA * HIDD + kk + half * 16;
    v4f a0 = *(const v4f*)(ap);
    v4f a1 = *(const v4f*)(ap + 4);
    v4f a2 = *(const v4f*)(ap + 8);
    v4f a3 = *(const v4f*)(ap + 12);
    v16bf amat;
#pragma unroll
    for (int i = 0; i < 4; ++i) {
      amat[i]      = (__bf16)a0[i];
      amat[4 + i]  = (__bf16)a1[i];
      amat[8 + i]  = (__bf16)a2[i];
      amat[12 + i] = (__bf16)a3[i];
    }
    // --- B tile (f32 -> bf16), W[n, kk+half*16 .. +15] contiguous ---
    const float* bp = W + (size_t)nB * HIDD + kk + half * 16;
    v4f b0 = *(const v4f*)(bp);
    v4f b1 = *(const v4f*)(bp + 4);
    v4f b2 = *(const v4f*)(bp + 8);
    v4f b3 = *(const v4f*)(bp + 12);
    v16bf bmat;
#pragma unroll
    for (int i = 0; i < 4; ++i) {
      bmat[i]      = (__bf16)b0[i];
      bmat[4 + i]  = (__bf16)b1[i];
      bmat[8 + i]  = (__bf16)b2[i];
      bmat[12 + i] = (__bf16)b3[i];
    }
    acc = __builtin_amdgcn_wmma_f32_16x16x32_bf16(false, amat, false, bmat,
                                                  (short)0, acc, false, false);
  }

  const int   n     = n0 + l16;
  const int   h     = n >> 6;         // n / HD
  const int   d     = n & 63;         // n % HD
  const float bval  = bias[n];
  const float scale = (which == 0) ? 0.125f : 1.0f;   // 1/sqrt(HD)=1/8 for q

#pragma unroll
  for (int r = 0; r < 8; ++r) {
    const int m    = m0 + r + half * 8;
    const int bidx = m >> 11;          // m / S
    const int s    = m & (SS - 1);     // m % S
    const __bf16 o = (__bf16)((acc[r] + bval) * scale);
    if (which == 2) {
      vT[(((size_t)bidx * NHH + h) * HDD + d) * SS + s] = o;   // [B,NH,HD,S]
    } else if (which == 0) {
      qb[(((size_t)bidx * NHH + h) * SS + s) * HDD + d] = o;   // [B,NH,S,HD]
    } else {
      kbuf[(((size_t)bidx * NHH + h) * SS + s) * HDD + d] = o;
    }
  }
}

// ---------------------------------------------------------------------------
// Kernel 2: flash attention. 4 waves / block, each wave owns a 16-row s-tile.
// Per 32-col t-chunk: 4 WMMAs (scores, K=64), bias+mask, online softmax,
// P restaged via LDS (C-layout f32 -> A-layout bf16), 4 WMMAs (P·V, K=32).
// ---------------------------------------------------------------------------
__global__ __launch_bounds__(128) void attn_kernel(
    const __bf16* __restrict__ qb, const __bf16* __restrict__ kbuf,
    const __bf16* __restrict__ vT,
    const float* __restrict__ rel, const float* __restrict__ rel2,
    const int* __restrict__ amask, float* __restrict__ out)
{
  __shared__ __align__(16) __bf16 pstage[4][16 * 32];

  const int lane = threadIdx.x & 31;
  const int wave = threadIdx.x >> 5;
  const int half = lane >> 4;
  const int l16  = lane & 15;

  const int s0 = (blockIdx.x * 4 + wave) * 16;
  const int h  = blockIdx.y;
  const int b  = blockIdx.z;

  const __bf16* qh  = qb   + ((size_t)b * NHH + h) * SS * HDD;
  const __bf16* kh  = kbuf + ((size_t)b * NHH + h) * SS * HDD;
  const __bf16* vh  = vT   + ((size_t)b * NHH + h) * HDD * SS;
  const float*  r1h = rel  + ((size_t)b * NHH + h) * SS * SS;
  const float*  r2h = rel2 + ((size_t)b * NHH + h) * SS * SS;
  const int*    mh  = amask + (size_t)b * SS * SS;

  // Q in A-operand form, two K=32 chunks over d (q already scaled by 1/8).
  v16bf aq[2];
#pragma unroll
  for (int c = 0; c < 2; ++c) {
    const __bf16* p = qh + (size_t)(s0 + l16) * HDD + c * 32 + half * 16;
    aq[c] = cat8(*(const v8bf*)p, *(const v8bf*)(p + 8));
  }

  v8f o[4] = {v8f{}, v8f{}, v8f{}, v8f{}};   // ctx accum 16x64
  float mi[8], li[8];
#pragma unroll
  for (int r = 0; r < 8; ++r) { mi[r] = FMIN; li[r] = 0.0f; }

  for (int t0 = 0; t0 < SS; t0 += 32) {
    // ---- scores: two 16x16 tiles over t (K = d = 64 via 2x K32 WMMA) ----
    v8f st[2];
#pragma unroll
    for (int tt = 0; tt < 2; ++tt) {
      const int t = t0 + tt * 16 + l16;
      const __bf16* kp = kh + (size_t)t * HDD + half * 16;
      v16bf bk0 = cat8(*(const v8bf*)(kp),      *(const v8bf*)(kp + 8));
      v16bf bk1 = cat8(*(const v8bf*)(kp + 32), *(const v8bf*)(kp + 40));
      v8f z = {};
      z = __builtin_amdgcn_wmma_f32_16x16x32_bf16(false, aq[0], false, bk0,
                                                  (short)0, z, false, false);
      st[tt] = __builtin_amdgcn_wmma_f32_16x16x32_bf16(false, aq[1], false, bk1,
                                                       (short)0, z, false, false);
    }

    // ---- bias + mask + online softmax (rows live in 16-lane halves) ----
#pragma unroll
    for (int r = 0; r < 8; ++r) {
      const int    srow   = s0 + r + half * 8;
      const size_t rowoff = (size_t)srow * SS + t0;

      __builtin_prefetch(r1h + rowoff + 32, 0, 1);   // global_prefetch_b8
      __builtin_prefetch(r2h + rowoff + 32, 0, 1);

      const int   mk0 = mh[rowoff + l16];
      const int   mk1 = mh[rowoff + 16 + l16];
      const float bb0 = r1h[rowoff + l16]      + r2h[rowoff + l16];
      const float bb1 = r1h[rowoff + 16 + l16] + r2h[rowoff + 16 + l16];
      float x0 = mk0 ? FMIN : st[0][r] + bb0;
      float x1 = mk1 ? FMIN : st[1][r] + bb1;

      float mx = fmaxf(x0, x1);
      mx = fmaxf(mx, __shfl_xor(mx, 1, 16));
      mx = fmaxf(mx, __shfl_xor(mx, 2, 16));
      mx = fmaxf(mx, __shfl_xor(mx, 4, 16));
      mx = fmaxf(mx, __shfl_xor(mx, 8, 16));

      const float mnew  = fmaxf(mi[r], mx);
      const float alpha = __expf(mi[r] - mnew);
      const float e0 = __expf(x0 - mnew);
      const float e1 = __expf(x1 - mnew);

      float rs = e0 + e1;
      rs += __shfl_xor(rs, 1, 16);
      rs += __shfl_xor(rs, 2, 16);
      rs += __shfl_xor(rs, 4, 16);
      rs += __shfl_xor(rs, 8, 16);

      li[r] = li[r] * alpha + rs;
      mi[r] = mnew;
#pragma unroll
      for (int dt = 0; dt < 4; ++dt) o[dt][r] *= alpha;

      // stage P (bf16) into LDS, row-major 16x32
      const int mrow = r + half * 8;
      pstage[wave][mrow * 32 + l16]      = (__bf16)e0;
      pstage[wave][mrow * 32 + 16 + l16] = (__bf16)e1;
    }

    asm volatile("s_wait_dscnt 0x0" ::: "memory");   // LDS RAW fence (in-wave)

    // re-read P in A-operand layout (16x32, K = t)
    const __bf16* pp = &pstage[wave][l16 * 32 + half * 16];
    const v16bf pa = cat8(*(const v8bf*)pp, *(const v8bf*)(pp + 8));

    // ---- P·V : 4 d-tiles of 16 cols, K = 32 t's, v stored transposed ----
#pragma unroll
    for (int dt = 0; dt < 4; ++dt) {
      const __bf16* vp = vh + (size_t)(dt * 16 + l16) * SS + t0 + half * 16;
      const v16bf bv16 = cat8(*(const v8bf*)vp, *(const v8bf*)(vp + 8));
      o[dt] = __builtin_amdgcn_wmma_f32_16x16x32_bf16(false, pa, false, bv16,
                                                      (short)0, o[dt], false, false);
    }
  }

  // ---- epilogue: normalize, write f32 [B,S,HID] ----
#pragma unroll
  for (int r = 0; r < 8; ++r) {
    const float inv  = 1.0f / li[r];
    const int   srow = s0 + r + half * 8;
    float* op = out + ((size_t)b * SS + srow) * HIDD + h * HDD;
#pragma unroll
    for (int dt = 0; dt < 4; ++dt) op[dt * 16 + l16] = o[dt][r] * inv;
  }
}

// ---------------------------------------------------------------------------
extern "C" void kernel_launch(void* const* d_in, const int* in_sizes, int n_in,
                              void* d_out, int out_size, void* d_ws, size_t ws_size,
                              hipStream_t stream) {
  const float* hs   = (const float*)d_in[0];
  const float* rel  = (const float*)d_in[1];
  const float* rel2 = (const float*)d_in[2];
  const int*   msk  = (const int*)d_in[3];
  const float* Wq   = (const float*)d_in[4];
  const float* bq   = (const float*)d_in[5];
  const float* Wk   = (const float*)d_in[6];
  const float* bk   = (const float*)d_in[7];
  const float* Wv   = (const float*)d_in[8];
  const float* bv   = (const float*)d_in[9];
  float* out = (float*)d_out;

  // workspace: q, k [B,NH,S,HD] bf16 ; vT [B,NH,HD,S] bf16  (~18 MB total)
  const size_t per = (size_t)BB * NHH * SS * HDD;
  __bf16* qb = (__bf16*)d_ws;
  __bf16* kb = qb + per;
  __bf16* vT = kb + per;

  // Kernel 1: M tiles = 4096/16 = 256 ; N tiles = 48 -> 6 groups of 8 waves ; z = q/k/v
  dim3 g1(256, 6, 3);
  qkv_proj_kernel<<<g1, 256, 0, stream>>>(hs, Wq, bq, Wk, bk, Wv, bv, qb, kb, vT);

  // Kernel 2: 4 waves/block, each owns 16 rows -> S/64 = 32 blocks per (h,b)
  dim3 g2(SS / 64, NHH, BB);
  attn_kernel<<<g2, 128, 0, stream>>>(qb, kb, vT, rel, rel2, msk, out);
}
